// AttentionBlock_59914793779507
// MI455X (gfx1250) — matrix operand order repaired and checked by reading, not compiled, not run on hardware
//
#include <hip/hip_runtime.h>
#include <hip/hip_bf16.h>

typedef __attribute__((ext_vector_type(16))) _Float16 v16h;
typedef __attribute__((ext_vector_type(8)))  _Float16 v8h;
typedef __attribute__((ext_vector_type(4)))  _Float16 v4h;
typedef __attribute__((ext_vector_type(8)))  float    v8f;
typedef __attribute__((ext_vector_type(4)))  float    v4f;

union HF16 { v16h v; v8h h2[2]; _Float16 h[16]; };

// Load 16 contiguous halfs (B-fragment: lane = column, 16 consecutive K values).
__device__ __forceinline__ v16h ld16h(const _Float16* p) {
  HF16 u;
  u.h2[0] = *(const v8h*)(p);
  u.h2[1] = *(const v8h*)(p + 8);
  return u.v;
}

// Load A-fragment (16x32 f16, MxK) from a row-major row pointer.
// Layout: lanes 0-15 hold K 0..15, lanes 16-31 hold K 16..31 (contiguous, matches B).
__device__ __forceinline__ v16h ldafrag(const _Float16* rowp, int lane) {
  int kb = (lane < 16) ? 0 : 16;
  HF16 u;
  u.h2[0] = *(const v8h*)(rowp + kb);
  u.h2[1] = *(const v8h*)(rowp + kb + 8);
  return u.v;
}

__device__ __forceinline__ v8f wmma_f16(v16h a, v16h b, v8f c) {
  return __builtin_amdgcn_wmma_f32_16x16x32_f16(false, a, false, b, (short)0, c,
                                                false, false);
}

#define BATCH 8
#define NTOK  4096   // H*W
#define CH    128
#define GRP   8
#define CPG   16     // CH / GRP
#define MTOT  (BATCH * NTOK)

// ---------------------------------------------------------------------------
// Kernel 1: GroupNorm. One block per (b, g); reduce 4096*16 elements.
// float4-vectorized. Writes fp32 (residual) and f16 (WMMA A-operand) copies.
// ---------------------------------------------------------------------------
__global__ __launch_bounds__(256)
void gn_kernel(const float* __restrict__ x, const float* __restrict__ gamma,
               const float* __restrict__ beta, float* __restrict__ xn32,
               _Float16* __restrict__ xn16) {
  int b = blockIdx.x >> 3;
  int g = blockIdx.x & 7;
  const size_t base = (size_t)b * NTOK * CH + g * CPG;
  const float* xb = x + base;

  float s = 0.f, s2 = 0.f;
  for (int i = threadIdx.x; i < NTOK * CPG / 4; i += 256) {
    int flat = i * 4;
    int sp = flat >> 4, ch = flat & 15;
    v4f v = *(const v4f*)(xb + (size_t)sp * CH + ch);
#pragma unroll
    for (int j = 0; j < 4; j++) { s += v[j]; s2 += v[j] * v[j]; }
  }
  __shared__ float rs[256], rq[256];
  rs[threadIdx.x] = s; rq[threadIdx.x] = s2;
  __syncthreads();
  for (int o = 128; o > 0; o >>= 1) {
    if ((int)threadIdx.x < o) {
      rs[threadIdx.x] += rs[threadIdx.x + o];
      rq[threadIdx.x] += rq[threadIdx.x + o];
    }
    __syncthreads();
  }
  const float inv_n = 1.0f / (float)(NTOK * CPG);
  float mean = rs[0] * inv_n;
  float var  = rq[0] * inv_n - mean * mean;
  float rstd = rsqrtf(var + 1e-5f);

  float* o32 = xn32 + base;
  _Float16* o16 = xn16 + base;
  for (int i = threadIdx.x; i < NTOK * CPG / 4; i += 256) {
    int flat = i * 4;
    int sp = flat >> 4, ch = flat & 15;
    size_t idx = (size_t)sp * CH + ch;
    v4f v = *(const v4f*)(xb + idx);
    v4f r; v4h rh;
#pragma unroll
    for (int j = 0; j < 4; j++) {
      int c = g * CPG + ch + j;
      float t = (v[j] - mean) * rstd * gamma[c] + beta[c];
      r[j] = t; rh[j] = (_Float16)t;
    }
    *(v4f*)(o32 + idx) = r;
    *(v4h*)(o16 + idx) = rh;
  }
}

// ---------------------------------------------------------------------------
// Kernel 2: projections (templated: 0=Q scaled, 1=K, 2=V stored transposed).
// Weight staged transposed in LDS as f16 so B-fragments are contiguous loads.
// ---------------------------------------------------------------------------
template <int WHICH>
__global__ __launch_bounds__(256)
void proj16_kernel(const _Float16* __restrict__ xn16, const float* __restrict__ w,
                   const float* __restrict__ bias, _Float16* __restrict__ dst) {
  __shared__ __align__(32) _Float16 wT[CH * CH];  // wT[n*128 + k] = w[k*128 + n]
  __shared__ float bias_s[CH];
  for (int i = threadIdx.x; i < CH * CH; i += 256) {
    int kk = i >> 7, n = i & 127;
    wT[n * CH + kk] = (_Float16)w[i];
  }
  if (threadIdx.x < CH) bias_s[threadIdx.x] = bias[threadIdx.x];
  __syncthreads();

  int wave = threadIdx.x >> 5, lane = threadIdx.x & 31;
  int lm = lane & 15, hi = lane >> 4;
  int kboff = hi ? 16 : 0;
  int m0 = blockIdx.x * 128 + wave * 16;

  const _Float16* arow = xn16 + (size_t)(m0 + lm) * CH;
  v16h a[4];
#pragma unroll
  for (int ks = 0; ks < 4; ks++) a[ks] = ldafrag(arow + ks * 32, lane);

  const float scale = (WHICH == 0) ? 0.08838834764831845f : 1.0f;  // 128^-0.5
#pragma unroll
  for (int nb = 0; nb < 8; nb++) {
    v16h bfr[4];
#pragma unroll
    for (int ks = 0; ks < 4; ks++)
      bfr[ks] = ld16h(&wT[(nb * 16 + lm) * CH + ks * 32 + kboff]);
    v8f acc = {};
#pragma unroll
    for (int ks = 0; ks < 4; ks++) acc = wmma_f16(a[ks], bfr[ks], acc);

    int col = nb * 16 + lm;
    float bval = bias_s[col];
    if (WHICH == 2) {
      // transposed store [B, C, N]: rows are consecutive tokens -> one v8h.
      v8h pk;
#pragma unroll
      for (int r = 0; r < 8; r++) pk[r] = (_Float16)(acc[r] + bval);
      int bidx = m0 >> 12;
      int tok0 = (m0 & (NTOK - 1)) + 8 * hi;
      *(v8h*)(dst + ((size_t)bidx * CH + col) * NTOK + tok0) = pk;
    } else {
#pragma unroll
      for (int r = 0; r < 8; r++) {
        int row = m0 + r + 8 * hi;
        dst[(size_t)row * CH + col] = (_Float16)((acc[r] + bval) * scale);
      }
    }
  }
}

// ---------------------------------------------------------------------------
// Kernel 3: Flash attention. grid = (NTOK/256, BATCH), 8 waves/block,
// 32 queries per wave (two A-tiles share every K/V B-fragment), keys in
// blocks of 32. All fragments of an iteration preloaded into distinct
// registers so loads overlap WMMAs instead of full-stalling on loadcnt 0.
// ---------------------------------------------------------------------------
__global__ __launch_bounds__(256)
void attn_kernel(const _Float16* __restrict__ qf, const _Float16* __restrict__ kf,
                 const _Float16* __restrict__ vtf, _Float16* __restrict__ of) {
  __shared__ __align__(32) _Float16 plds[8][2][16 * 32];  // wave-private P tiles

  int wave = threadIdx.x >> 5, lane = threadIdx.x & 31;
  int lm = lane & 15, hi = lane >> 4;
  int kboff = hi ? 16 : 0;
  int b = blockIdx.y;
  int q0 = blockIdx.x * 256 + wave * 32;

  const _Float16* qb  = qf  + ((size_t)b * NTOK + q0) * CH;
  const _Float16* kbp = kf  + (size_t)b * NTOK * CH;
  const _Float16* vtb = vtf + (size_t)b * CH * NTOK;

  v16h aq[2][4];
#pragma unroll
  for (int t = 0; t < 2; t++)
#pragma unroll
    for (int ks = 0; ks < 4; ks++)
      aq[t][ks] = ldafrag(qb + (size_t)(t * 16 + lm) * CH + ks * 32, lane);

  v8f o[2][8] = {};
  float mi[2][8], li[2][8];
#pragma unroll
  for (int t = 0; t < 2; t++)
#pragma unroll
    for (int r = 0; r < 8; r++) { mi[t][r] = -3.0e38f; li[t][r] = 0.f; }

  for (int j0 = 0; j0 < NTOK; j0 += 32) {
    // Prefetch next key/value block into the cache hierarchy.
    __builtin_prefetch(kbp + (size_t)(j0 + 32 + lm) * CH + kboff, 0, 1);
    __builtin_prefetch(vtb + (size_t)lm * NTOK + (j0 + 32) + kboff, 0, 1);

    // Preload all K and V fragments of this iteration (distinct registers).
    v16h kb0[4], kb1[4];
#pragma unroll
    for (int ks = 0; ks < 4; ks++) {
      kb0[ks] = ld16h(kbp + (size_t)(j0 + lm) * CH + ks * 32 + kboff);
      kb1[ks] = ld16h(kbp + (size_t)(j0 + 16 + lm) * CH + ks * 32 + kboff);
    }
    v16h bv[8];
#pragma unroll
    for (int cb = 0; cb < 8; cb++)
      bv[cb] = ld16h(vtb + (size_t)(cb * 16 + lm) * NTOK + j0 + kboff);

    // S = Q . K^T : 2 query tiles x 2 key tiles, K-reduction over C=128.
    // Four independent accumulator chains hide WMMA->WMMA hazards.
    v8f s[2][2] = {};
#pragma unroll
    for (int ks = 0; ks < 4; ks++) {
      s[0][0] = wmma_f16(aq[0][ks], kb0[ks], s[0][0]);
      s[1][0] = wmma_f16(aq[1][ks], kb0[ks], s[1][0]);
      s[0][1] = wmma_f16(aq[0][ks], kb1[ks], s[0][1]);
      s[1][1] = wmma_f16(aq[1][ks], kb1[ks], s[1][1]);
    }
    // Online softmax: rows live in VGPRs, columns across each 16-lane half.
#pragma unroll
    for (int t = 0; t < 2; t++) {
      _Float16* pl = &plds[wave][t][0];
#pragma unroll
      for (int r = 0; r < 8; r++) {
        float mx = fmaxf(s[t][0][r], s[t][1][r]);
        mx = fmaxf(mx, __shfl_xor(mx, 1));
        mx = fmaxf(mx, __shfl_xor(mx, 2));
        mx = fmaxf(mx, __shfl_xor(mx, 4));
        mx = fmaxf(mx, __shfl_xor(mx, 8));
        float mnew  = fmaxf(mi[t][r], mx);
        float alpha = __expf(mi[t][r] - mnew);
        float p0 = __expf(s[t][0][r] - mnew);
        float p1 = __expf(s[t][1][r] - mnew);
        float rsum = p0 + p1;
        rsum += __shfl_xor(rsum, 1);
        rsum += __shfl_xor(rsum, 2);
        rsum += __shfl_xor(rsum, 4);
        rsum += __shfl_xor(rsum, 8);
        li[t][r] = li[t][r] * alpha + rsum;
        mi[t][r] = mnew;
#pragma unroll
        for (int cb = 0; cb < 8; cb++) o[t][cb][r] *= alpha;
        // C-layout -> row-major P tile in LDS (row = r + 8*hi)
        pl[(r + 8 * hi) * 32 + lm]      = (_Float16)p0;
        pl[(r + 8 * hi) * 32 + 16 + lm] = (_Float16)p1;
      }
    }
    // Reload P tiles as A-fragments (wave-private LDS; DS ops are in order).
    v16h ap0 = ldafrag(&plds[wave][0][lm * 32], lane);
    v16h ap1 = ldafrag(&plds[wave][1][lm * 32], lane);
    // P . V : 16 independent WMMAs against preloaded V fragments.
#pragma unroll
    for (int cb = 0; cb < 8; cb++) {
      o[0][cb] = wmma_f16(ap0, bv[cb], o[0][cb]);
      o[1][cb] = wmma_f16(ap1, bv[cb], o[1][cb]);
    }
  }

#pragma unroll
  for (int t = 0; t < 2; t++)
#pragma unroll
    for (int r = 0; r < 8; r++) {
      float inv = 1.0f / li[t][r];
      int row = q0 + t * 16 + r + 8 * hi;
      _Float16* orow = of + ((size_t)b * NTOK + row) * CH;
#pragma unroll
      for (int cb = 0; cb < 8; cb++)
        orow[cb * 16 + lm] = (_Float16)(o[t][cb][r] * inv);
    }
}

// ---------------------------------------------------------------------------
// Kernel 4: output projection + bias + residual (normalized input), fp32 out.
// ---------------------------------------------------------------------------
__global__ __launch_bounds__(256)
void out_kernel(const _Float16* __restrict__ of, const float* __restrict__ wp,
                const float* __restrict__ bp, const float* __restrict__ xn32,
                float* __restrict__ out) {
  __shared__ __align__(32) _Float16 wT[CH * CH];
  __shared__ float bias_s[CH];
  for (int i = threadIdx.x; i < CH * CH; i += 256) {
    int kk = i >> 7, n = i & 127;
    wT[n * CH + kk] = (_Float16)wp[i];
  }
  if (threadIdx.x < CH) bias_s[threadIdx.x] = bp[threadIdx.x];
  __syncthreads();

  int wave = threadIdx.x >> 5, lane = threadIdx.x & 31;
  int lm = lane & 15, hi = lane >> 4;
  int kboff = hi ? 16 : 0;
  int m0 = blockIdx.x * 128 + wave * 16;

  const _Float16* arow = of + (size_t)(m0 + lm) * CH;
  v16h a[4];
#pragma unroll
  for (int ks = 0; ks < 4; ks++) a[ks] = ldafrag(arow + ks * 32, lane);

#pragma unroll
  for (int nb = 0; nb < 8; nb++) {
    v16h bfr[4];
#pragma unroll
    for (int ks = 0; ks < 4; ks++)
      bfr[ks] = ld16h(&wT[(nb * 16 + lm) * CH + ks * 32 + kboff]);
    v8f acc = {};
#pragma unroll
    for (int ks = 0; ks < 4; ks++) acc = wmma_f16(a[ks], bfr[ks], acc);

    int col = nb * 16 + lm;
    float bval = bias_s[col];
#pragma unroll
    for (int r = 0; r < 8; r++) {
      int row = m0 + r + 8 * hi;
      size_t idx = (size_t)row * CH + col;
      out[idx] = acc[r] + bval + xn32[idx];
    }
  }
}

extern "C" void kernel_launch(void* const* d_in, const int* in_sizes, int n_in,
                              void* d_out, int out_size, void* d_ws, size_t ws_size,
                              hipStream_t stream) {
  const float* x     = (const float*)d_in[0];
  const float* gamma = (const float*)d_in[1];
  const float* beta  = (const float*)d_in[2];
  const float* wq = (const float*)d_in[3];
  const float* bq = (const float*)d_in[4];
  const float* wk = (const float*)d_in[5];
  const float* bk = (const float*)d_in[6];
  const float* wv = (const float*)d_in[7];
  const float* bv = (const float*)d_in[8];
  const float* wp = (const float*)d_in[9];
  const float* bp = (const float*)d_in[10];
  float* out = (float*)d_out;

  char* ws = (char*)d_ws;
  const size_t F32_BYTES = (size_t)MTOT * CH * sizeof(float);     // 16 MiB
  const size_t F16_BYTES = (size_t)MTOT * CH * sizeof(_Float16);  // 8 MiB
  float*    xn32 = (float*)ws;
  _Float16* xn16 = (_Float16*)(ws + F32_BYTES);
  _Float16* qf   = (_Float16*)(ws + F32_BYTES + 1 * F16_BYTES);
  _Float16* kf   = (_Float16*)(ws + F32_BYTES + 2 * F16_BYTES);
  _Float16* vtf  = (_Float16*)(ws + F32_BYTES + 3 * F16_BYTES);
  _Float16* of   = (_Float16*)(ws + F32_BYTES + 4 * F16_BYTES);

  gn_kernel<<<BATCH * GRP, 256, 0, stream>>>(x, gamma, beta, xn32, xn16);
  proj16_kernel<0><<<MTOT / 128, 256, 0, stream>>>(xn16, wq, bq, qf);
  proj16_kernel<1><<<MTOT / 128, 256, 0, stream>>>(xn16, wk, bk, kf);
  proj16_kernel<2><<<MTOT / 128, 256, 0, stream>>>(xn16, wv, bv, vtf);
  attn_kernel<<<dim3(NTOK / 256, BATCH), 256, 0, stream>>>(qf, kf, vtf, of);
  out_kernel<<<MTOT / 128, 256, 0, stream>>>(of, wp, bp, xn32, out);
}
